// RelationExtractor_86663850098726
// MI455X (gfx1250) — compile-verified, matrix operand order checked by reading, not run on hardware
//
#include <hip/hip_runtime.h>
#include <hip/hip_bf16.h>

#define D_IN  768
#define D_H   128
#define NG    512   // 4*D_H gate width
#define SEQ   512
#define NBATCH 64

typedef __attribute__((ext_vector_type(16))) _Float16 v16h;
typedef __attribute__((ext_vector_type(8)))  _Float16 v8h;
typedef __attribute__((ext_vector_type(4)))  _Float16 v4h;
typedef __attribute__((ext_vector_type(8)))  float    v8f;

union FragH {
    v16h v;
    v8h  h2[2];
    _Float16 e[16];
};

// sigmoid via hardware exp + rcp (TRANS pipe, co-executes with VALU/WMMA)
__device__ __forceinline__ float sigmoid_fast(float x) {
    return __builtin_amdgcn_rcpf(1.0f + __expf(-x));
}
// tanh(x) = (1 - e^{-2x}) / (1 + e^{-2x})
__device__ __forceinline__ float tanh_fast(float x) {
    const float e = __expf(-2.0f * x);
    return (1.0f - e) * __builtin_amdgcn_rcpf(1.0f + e);
}

// ---------------------------------------------------------------------------
// Kernel 0: bulk fp32 -> f16 conversion (vectorized float4 -> 4xf16)
// ---------------------------------------------------------------------------
__global__ __launch_bounds__(256)
void cvt_f32_f16_kernel(const float* __restrict__ src, _Float16* __restrict__ dst,
                        int n4) {
    const int i = blockIdx.x * 256 + threadIdx.x;
    if (i < n4) {
        float4 v = ((const float4*)src)[i];
        v4h o = { (_Float16)v.x, (_Float16)v.y, (_Float16)v.z, (_Float16)v.w };
        ((v4h*)dst)[i] = o;
    }
}

// ---------------------------------------------------------------------------
// Kernel 1: input projection + biases, hoisted out of the scan.
// G is stored GATE-INTERLEAVED: G[row][j*4 + gate], gate order (i,f,g,o),
// so the recurrence can fetch all 4 gate pre-activations of unit j with one
// aligned 8-byte load.
// Block = 32 rows, 8 waves; wave owns a 64-col strip, 2 M-tiles x 4 N-tiles.
// ---------------------------------------------------------------------------
__global__ __launch_bounds__(256)
void lstm_input_gemm_kernel(const _Float16* __restrict__ Xh,    // [32768, 768]
                            const _Float16* __restrict__ Wih_h, // [512, 768]
                            const float* __restrict__ bih,      // [512]
                            const float* __restrict__ bhh,      // [512]
                            _Float16* __restrict__ G)           // [32768, 512] interleaved
{
    const int lane  = threadIdx.x & 31;
    const int wave  = threadIdx.x >> 5;
    const int row0  = blockIdx.x * 32;
    const int col0  = wave * 64;
    const int laneM = lane & 15;
    const int koff  = (lane >> 4) * 8;
    const int mbase = (lane >> 4) * 8;

    v8f acc[2][4] = {};

    const _Float16* xr0 = Xh + (size_t)(row0 + laneM) * D_IN;
    const _Float16* xr1 = xr0 + (size_t)16 * D_IN;

    for (int k0 = 0; k0 < D_IN; k0 += 32) {
        FragH a0, a1;
        a0.h2[0] = *(const v8h*)(xr0 + k0 + koff);
        a0.h2[1] = *(const v8h*)(xr0 + k0 + 16 + koff);
        a1.h2[0] = *(const v8h*)(xr1 + k0 + koff);
        a1.h2[1] = *(const v8h*)(xr1 + k0 + 16 + koff);
#pragma unroll
        for (int nt = 0; nt < 4; ++nt) {
            const int col = col0 + nt * 16 + laneM;
            const _Float16* wr = Wih_h + (size_t)col * D_IN + k0 + koff;
            FragH b;
            b.h2[0] = *(const v8h*)wr;
            b.h2[1] = *(const v8h*)(wr + 16);
            acc[0][nt] = __builtin_amdgcn_wmma_f32_16x16x32_f16(
                false, a0.v, false, b.v, (short)0, acc[0][nt], false, false);
            acc[1][nt] = __builtin_amdgcn_wmma_f32_16x16x32_f16(
                false, a1.v, false, b.v, (short)0, acc[1][nt], false, false);
        }
    }

#pragma unroll
    for (int nt = 0; nt < 4; ++nt) {
        const int col  = col0 + nt * 16 + laneM;
        const int gate = col >> 7;      // 0..3 (i,f,g,o)
        const int jj   = col & 127;     // hidden unit
        const float bias = bih[col] + bhh[col];
#pragma unroll
        for (int mt = 0; mt < 2; ++mt) {
            _Float16* g = G + (size_t)(row0 + mt * 16 + mbase) * NG + jj * 4 + gate;
#pragma unroll
            for (int r = 0; r < 8; ++r) {
                g[(size_t)r * NG] = (_Float16)(acc[mt][nt][r] + bias);
            }
        }
    }
}

// ---------------------------------------------------------------------------
// Kernel 2: serial LSTM recurrence. 4 blocks x 16 documents each.
// Per step: 8 ds_load_b128 (h frags) -> 16 WMMA (W_hh in registers) ->
// interleaved spill -> barrier -> one ds_load_b128 + one 8B global load per
// element + exp/rcp nonlinearity -> barrier.
// ---------------------------------------------------------------------------
__global__ __launch_bounds__(256)
void lstm_recurrence_kernel(const _Float16* __restrict__ G,     // [B*S, 512] interleaved f16
                            const _Float16* __restrict__ Whh_h, // [512, 128] f16
                            const float* __restrict__ Wcls,     // [128]
                            const float* __restrict__ bcls,     // [1]
                            float* __restrict__ out)            // [64]
{
    __shared__ __align__(16) float gatesS[16 * NG];     // 32 KB, [m][j*4+gate]
    __shared__ float cS[16 * D_H];                      //  8 KB
    __shared__ __align__(16) _Float16 hhalf[16 * D_H];  //  4 KB

    const int tid   = threadIdx.x;
    const int lane  = tid & 31;
    const int wave  = tid >> 5;
    const int b0    = blockIdx.x * 16;
    const int laneM = lane & 15;
    const int koff  = (lane >> 4) * 8;
    const int mbase = (lane >> 4) * 8;

    // ---- hoist entire W_hh slice for this wave into registers ----
    FragH bw[4][4];   // [kk][nt] : 16 fragments = 128 VGPRs
#pragma unroll
    for (int kk = 0; kk < 4; ++kk) {
#pragma unroll
        for (int nt = 0; nt < 4; ++nt) {
            const int col = wave * 64 + nt * 16 + laneM;
            const _Float16* wr = Whh_h + (size_t)col * D_H + kk * 32 + koff;
            bw[kk][nt].h2[0] = *(const v8h*)wr;
            bw[kk][nt].h2[1] = *(const v8h*)(wr + 16);
        }
    }

    for (int i = tid; i < 16 * D_H; i += 256) {
        cS[i] = 0.0f;
        hhalf[i] = (_Float16)0.0f;
    }
    __syncthreads();

    for (int t = 0; t < SEQ; ++t) {
        // ---- h @ W_hh^T : A-fragments from LDS, B from registers ----
        v8f acc[4] = {};
#pragma unroll
        for (int kk = 0; kk < 4; ++kk) {
            const int k0 = kk * 32;
            FragH a;
            a.h2[0] = *(const v8h*)&hhalf[laneM * D_H + k0 + koff];
            a.h2[1] = *(const v8h*)&hhalf[laneM * D_H + k0 + 16 + koff];
#pragma unroll
            for (int nt = 0; nt < 4; ++nt) {
                acc[nt] = __builtin_amdgcn_wmma_f32_16x16x32_f16(
                    false, a.v, false, bw[kk][nt].v, (short)0, acc[nt], false, false);
            }
        }

        // ---- spill recurrent contribution to LDS (gate-interleaved) ----
#pragma unroll
        for (int nt = 0; nt < 4; ++nt) {
            const int col  = wave * 64 + nt * 16 + laneM;
            const int gate = col >> 7;
            const int jj   = col & 127;
#pragma unroll
            for (int r = 0; r < 8; ++r) {
                gatesS[(mbase + r) * NG + jj * 4 + gate] = acc[nt][r];
            }
        }
        __syncthreads();

        // ---- fuse Gpre (one 8B load) + cell nonlinearity (TRANS pipe) ----
#pragma unroll
        for (int e = 0; e < 8; ++e) {
            const int idx = tid + e * 256;   // 0..2047
            const int m = idx >> 7;
            const int j = idx & 127;
            const float4 gr = *(const float4*)&gatesS[m * NG + j * 4];
            const _Float16* gp = G + ((size_t)(b0 + m) * SEQ + t) * NG + j * 4;
            const v4h gv = *(const v4h*)gp;
            const float ig = gr.x + (float)gv[0];
            const float fg = gr.y + (float)gv[1];
            const float gg = gr.z + (float)gv[2];
            const float og = gr.w + (float)gv[3];
            const float i_s = sigmoid_fast(ig);
            const float f_s = sigmoid_fast(fg);
            const float o_s = sigmoid_fast(og);
            const float g_t = tanh_fast(gg);
            const float c_new = f_s * cS[m * D_H + j] + i_s * g_t;
            cS[m * D_H + j] = c_new;
            const float h_new = o_s * tanh_fast(c_new);
            hhalf[m * D_H + j] = (_Float16)h_new;
            // sparse prefetch of next step's pre-activations (next row, +NG)
            if ((t + 1 < SEQ) && ((idx & 31) == 0)) {
                __builtin_prefetch(gp + NG, 0, 0);
            }
        }
        __syncthreads();
    }

    // ---- classifier on final hidden state ----
    if (tid < 16) {
        float s = 0.0f;
#pragma unroll 4
        for (int j = 0; j < D_H; ++j) s += (float)hhalf[tid * D_H + j] * Wcls[j];
        out[b0 + tid] = s + bcls[0];
    }
}

extern "C" void kernel_launch(void* const* d_in, const int* in_sizes, int n_in,
                              void* d_out, int out_size, void* d_ws, size_t ws_size,
                              hipStream_t stream) {
    const float* feat = (const float*)d_in[0];  // [64, 512, 768]
    const float* Wih  = (const float*)d_in[1];  // [512, 768]
    const float* Whh  = (const float*)d_in[2];  // [512, 128]
    const float* bih  = (const float*)d_in[3];  // [512]
    const float* bhh  = (const float*)d_in[4];  // [512]
    const float* Wcls = (const float*)d_in[5];  // [1, 128]
    const float* bcls = (const float*)d_in[6];  // [1]
    float* out = (float*)d_out;                 // [64, 1]

    // workspace layout (f16):
    //   G    : [32768, 512]  32 MB   @ 0        (gate-interleaved)
    //   Xh   : [32768, 768]  48 MB   @ 32 MB
    //   WihH : [512, 768]   0.75 MB  @ 80 MB
    //   WhhH : [512, 128]  0.125 MB  @ 80.75 MB
    char* ws = (char*)d_ws;
    _Float16* G    = (_Float16*)(ws);
    _Float16* Xh   = (_Float16*)(ws + (size_t)33554432);
    _Float16* WihH = (_Float16*)(ws + (size_t)33554432 + 50331648);
    _Float16* WhhH = (_Float16*)(ws + (size_t)33554432 + 50331648 + 786432);

    // one-time f16 conversions (bandwidth-trivial)
    {
        const int nX4 = (NBATCH * SEQ * D_IN) / 4;   // 6291456
        cvt_f32_f16_kernel<<<(nX4 + 255) / 256, 256, 0, stream>>>(feat, Xh, nX4);
        const int nWih4 = (NG * D_IN) / 4;           // 98304
        cvt_f32_f16_kernel<<<(nWih4 + 255) / 256, 256, 0, stream>>>(Wih, WihH, nWih4);
        const int nWhh4 = (NG * D_H) / 4;            // 16384
        cvt_f32_f16_kernel<<<(nWhh4 + 255) / 256, 256, 0, stream>>>(Whh, WhhH, nWhh4);
    }

    // Big parallel GEMM: all input projections + biases, hoisted out of scan.
    lstm_input_gemm_kernel<<<(NBATCH * SEQ) / 32, 256, 0, stream>>>(
        Xh, WihH, bih, bhh, G);

    // Serial recurrence, batch-parallel across 4 workgroups.
    lstm_recurrence_kernel<<<NBATCH / 16, 256, 0, stream>>>(
        G, WhhH, Wcls, bcls, out);
}
